// VectorQuantizer_49675591746039
// MI455X (gfx1250) — matrix operand order repaired and checked by reading, not compiled, not run on hardware
//
#include <hip/hip_runtime.h>
#include <hip/hip_bf16.h>

// ---------------------------------------------------------------------------
// VQ forward for MI455X (gfx1250, wave32, WMMA).
// Distances via split-bf16 WMMA (Ah*Bh + Ah*Bl + Al*Bh) for ~fp32 accuracy
// at bf16 matrix-core rate. Codebook (2MB) is L2-resident (192MB L2), so the
// main loop streams packed fragments straight from global memory, with
// WGP-scope (default-scope) global_prefetch_b8 of the next tile via inline
// asm. Two independent WMMA accumulator chains per tile halve the RAW
// dependency depth on the matrix pipeline.
// ---------------------------------------------------------------------------

typedef __bf16 bf16;
typedef __attribute__((ext_vector_type(16))) __bf16 v16bf;
typedef __attribute__((ext_vector_type(8)))  float  v8f;

#define N_ROWS   16384      // 8*2048
#define DIM      64
#define K_EMB    8192
#define TOPK     10
#define ROW_TILES (N_ROWS/16)   // 1024
#define CODE_TILES (K_EMB/16)   // 512

// d_out layout (floats): quantized[N*D], total_loss, idx[N], min_d[N], perplexity
#define Q_OFF    0
#define TL_OFF   (N_ROWS*DIM)                 // 1048576
#define IDX_OFF  (TL_OFF+1)                   // 1048577
#define MD_OFF   (IDX_OFF+N_ROWS)             // 1064961
#define PERP_OFF (MD_OFF+N_ROWS)              // 1081345

// ---------------------------------------------------------------------------
// init: zero counts + scalar accumulators
// ---------------------------------------------------------------------------
__global__ void vq_init(float* __restrict__ counts, float* __restrict__ scalars) {
    int i = blockIdx.x * blockDim.x + threadIdx.x;
    if (i < K_EMB) counts[i] = 0.0f;
    if (i < 2)     scalars[i] = 0.0f;
}

// ---------------------------------------------------------------------------
// stats: per-row sum of squares for inputs (xx) and codebook (ee)
// ---------------------------------------------------------------------------
__global__ void vq_stats(const float* __restrict__ inp, const float* __restrict__ cb,
                         float* __restrict__ xx, float* __restrict__ ee) {
    int i = blockIdx.x * blockDim.x + threadIdx.x;
    const float* src;
    float* dst;
    if (i < N_ROWS) {
        src = inp + (size_t)i * DIM; dst = xx + i;
    } else {
        int r = i - N_ROWS;
        if (r >= K_EMB) return;
        src = cb + (size_t)r * DIM; dst = ee + r;
    }
    float s = 0.0f;
    const float4* p = (const float4*)src;
#pragma unroll
    for (int j = 0; j < DIM / 4; ++j) {
        float4 v = p[j];
        s += v.x * v.x + v.y * v.y + v.z * v.z + v.w * v.w;
    }
    *dst = s;
}

// ---------------------------------------------------------------------------
// pack: fp32 [16-row tiles x 64] -> wave32 WMMA fragments, split into
// bf16 hi + bf16 lo (residual).  isA=1 -> A-matrix K-interleave layout,
// isA=0 -> B-matrix, same K-interleave layout.  One thread per (tile,kblock,lane).
// Fragment memory: ((tile*2 + kb)*32 + lane) * 16 bf16  (32B per lane).
// ---------------------------------------------------------------------------
__global__ void vq_pack(const float* __restrict__ src, bf16* __restrict__ hi,
                        bf16* __restrict__ lo, int numTiles, int isA) {
    int gid = blockIdx.x * blockDim.x + threadIdx.x;
    if (gid >= numTiles * 2 * 32) return;
    int lane = gid & 31;
    int kb   = (gid >> 5) & 1;
    int t    = gid >> 6;
    int row  = t * 16 + (lane & 15);
    const float* s = src + (size_t)row * DIM + kb * 32;

    v16bf vh, vl;
#pragma unroll
    for (int j = 0; j < 16; ++j) {
        int k;
        if (isA) k = ((lane & 16) >> 1) + (j < 8 ? j : j + 8);  // {0-7,16-23} / {8-15,24-31}
        else     k = ((lane & 16) >> 1) + (j < 8 ? j : j + 8);  // {0-7,16-23} / {8-15,24-31}
        float v  = s[k];
        bf16 h   = (bf16)v;
        vh[j] = h;
        vl[j] = (bf16)(v - (float)h);
    }
    ((v16bf*)hi)[gid] = vh;
    ((v16bf*)lo)[gid] = vl;
}

// branchless sorted insert into ascending 10-element register arrays
#define TOPK_INSERT(dv, iv)                                              \
    {                                                                    \
        float _d = (dv); int _i = (iv);                                  \
        float _nd[TOPK]; int _ni[TOPK];                                  \
        _Pragma("unroll")                                                \
        for (int _s = TOPK - 1; _s >= 0; --_s) {                         \
            int  _sp  = _s > 0 ? _s - 1 : 0;                             \
            bool _lt  = _d < best_d[_s];                                 \
            bool _ltp = (_s > 0) && (_d < best_d[_sp]);                  \
            _nd[_s] = _lt ? (_ltp ? best_d[_sp] : _d) : best_d[_s];      \
            _ni[_s] = _lt ? (_ltp ? best_i[_sp] : _i) : best_i[_s];      \
        }                                                                \
        _Pragma("unroll")                                                \
        for (int _s = 0; _s < TOPK; ++_s) {                              \
            best_d[_s] = _nd[_s]; best_i[_s] = _ni[_s];                  \
        }                                                                \
    }

// ---------------------------------------------------------------------------
// main: one wave per 16-row tile. Streams all 512 code tiles, 6 WMMAs each
// (split-bf16, two independent accumulator chains), fused top-10 ->
// gumbel-max sample -> gather -> outputs.
// ---------------------------------------------------------------------------
__global__ __launch_bounds__(256) void vq_main(
    const float* __restrict__ inp, const float* __restrict__ cb,
    const float* __restrict__ gum,
    const bf16* __restrict__ inHi, const bf16* __restrict__ inLo,
    const bf16* __restrict__ cbHi, const bf16* __restrict__ cbLo,
    const float* __restrict__ xx, const float* __restrict__ ee,
    float* __restrict__ counts, float* __restrict__ scalars,
    float* __restrict__ out) {

    const int lane    = threadIdx.x & 31;
    const int wave    = threadIdx.x >> 5;
    const int rowTile = blockIdx.x * 8 + wave;
    const int rowBase = rowTile * 16;
    const int myRow   = rowBase + (lane & 15);
    const int mOff    = (lane & 16) >> 1;        // lanes>=16 own codes m=8..15

    const float xxv = xx[myRow];

    // resident B fragments (this wave's 16 rows), hi/lo x two K-blocks
    const v16bf* inHiV = (const v16bf*)inHi;
    const v16bf* inLoV = (const v16bf*)inLo;
    const v16bf Bh0 = inHiV[(rowTile * 2 + 0) * 32 + lane];
    const v16bf Bh1 = inHiV[(rowTile * 2 + 1) * 32 + lane];
    const v16bf Bl0 = inLoV[(rowTile * 2 + 0) * 32 + lane];
    const v16bf Bl1 = inLoV[(rowTile * 2 + 1) * 32 + lane];

    float best_d[TOPK];
    int   best_i[TOPK];
#pragma unroll
    for (int s = 0; s < TOPK; ++s) { best_d[s] = 3.4e38f; best_i[s] = 0; }

    const v16bf* cbHiV = (const v16bf*)cbHi;
    const v16bf* cbLoV = (const v16bf*)cbLo;

    for (int t = 0; t < CODE_TILES; ++t) {
        v16bf Ah0 = cbHiV[(t * 2 + 0) * 32 + lane];
        v16bf Ah1 = cbHiV[(t * 2 + 1) * 32 + lane];
        v16bf Al0 = cbLoV[(t * 2 + 0) * 32 + lane];
        v16bf Al1 = cbLoV[(t * 2 + 1) * 32 + lane];

        // Pull next tile's fragments into the near (WGP) cache while WMMAs
        // run. Default scope (no scope: modifier) == SCOPE 0 (WGP), which
        // fills every cache level on a miss -- __builtin_prefetch can only
        // express SE/SYS scopes, which skip the WGP cache, so use inline asm.
        // Speculative prefetch past the end at t=511 is silently dropped.
        {
            const v16bf* pfHi = &cbHiV[(t * 2 + 2) * 32 + lane];
            const v16bf* pfLo = &cbLoV[(t * 2 + 2) * 32 + lane];
            asm volatile("global_prefetch_b8 %0, off" :: "v"(pfHi));
            asm volatile("global_prefetch_b8 %0, off" :: "v"(pfLo));
        }

        // two independent accumulator chains (K-block 0 / K-block 1)
        v8f acc0 = {};
        v8f acc1 = {};
        acc0 = __builtin_amdgcn_wmma_f32_16x16x32_bf16(false, Ah0, false, Bh0, (short)0, acc0, false, false);
        acc1 = __builtin_amdgcn_wmma_f32_16x16x32_bf16(false, Ah1, false, Bh1, (short)0, acc1, false, false);
        acc0 = __builtin_amdgcn_wmma_f32_16x16x32_bf16(false, Ah0, false, Bl0, (short)0, acc0, false, false);
        acc1 = __builtin_amdgcn_wmma_f32_16x16x32_bf16(false, Ah1, false, Bl1, (short)0, acc1, false, false);
        acc0 = __builtin_amdgcn_wmma_f32_16x16x32_bf16(false, Al0, false, Bh0, (short)0, acc0, false, false);
        acc1 = __builtin_amdgcn_wmma_f32_16x16x32_bf16(false, Al1, false, Bh1, (short)0, acc1, false, false);

        v8f eev = *(const v8f*)(ee + t * 16 + mOff);
        float kth = best_d[TOPK - 1];
#pragma unroll
        for (int j = 0; j < 8; ++j) {
            float dist = xxv + eev[j] - 2.0f * (acc0[j] + acc1[j]);
            if (dist < kth) {
                TOPK_INSERT(dist, t * 16 + mOff + j);
                kth = best_d[TOPK - 1];
            }
        }
    }

    // merge partner half-wave's list (disjoint code sets, both sorted)
    float pd[TOPK]; int pi[TOPK];
#pragma unroll
    for (int j = 0; j < TOPK; ++j) {
        pd[j] = __shfl_xor(best_d[j], 16);
        pi[j] = __shfl_xor(best_i[j], 16);
    }
#pragma unroll
    for (int j = 0; j < TOPK; ++j) {
        if (pd[j] < best_d[TOPK - 1]) TOPK_INSERT(pd[j], pi[j]);
    }

    // gumbel-max sample over top-10 (TEMP = 1)
    const float validf = (xxv > 1e-12f) ? 1.0f : 0.0f;
    int selIdx = 0; float selD = 0.0f;
    {
        float bestv = -3.4e38f;
#pragma unroll
        for (int j = 0; j < TOPK; ++j) {
            float lg = -best_d[j] + gum[(size_t)myRow * TOPK + j];
            if (lg > bestv) { bestv = lg; selIdx = best_i[j]; selD = best_d[j]; }
        }
    }
    if (validf == 0.0f) { selIdx = 0; selD = 0.0f; }

    if (lane < 16) {
        out[IDX_OFF + myRow] = (float)selIdx;
        out[MD_OFF + myRow]  = selD;
        if (validf != 0.0f) atomicAdd(&counts[selIdx], 1.0f);
    }

    // cooperative gather of selected codebook rows + vq-loss accumulation
    float lsum = 0.0f;
#pragma unroll 1
    for (int r = 0; r < 16; ++r) {
        int   rowR = rowBase + r;
        int   idxR = __shfl(selIdx, r);
        float vR   = __shfl(validf, r);
        const float* crow = cb + (size_t)idxR * DIM;
        float c0 = crow[lane], c1 = crow[lane + 32];
        size_t o = (size_t)rowR * DIM;
        out[Q_OFF + o + lane]      = vR * c0;   // quantized_st forward == q*mask
        out[Q_OFF + o + lane + 32] = vR * c1;
        float d0 = c0 - inp[o + lane];
        float d1 = c1 - inp[o + lane + 32];
        lsum += vR * (d0 * d0 + d1 * d1);
    }

    float vcontrib = (lane < 16) ? validf : 0.0f;
#pragma unroll
    for (int off = 16; off > 0; off >>= 1) {
        lsum     += __shfl_xor(lsum, off);
        vcontrib += __shfl_xor(vcontrib, off);
    }
    if (lane == 0) {
        atomicAdd(&scalars[1], lsum);      // sum vmask*(q-x)^2
        atomicAdd(&scalars[0], vcontrib);  // n_valid
    }
}

// ---------------------------------------------------------------------------
// finalize: perplexity + total loss scalars
// ---------------------------------------------------------------------------
__global__ void vq_finalize(const float* __restrict__ counts,
                            const float* __restrict__ scalars,
                            float* __restrict__ out) {
    __shared__ float red[256];
    int tid = threadIdx.x;
    float nv = fmaxf(scalars[0], 1.0f);
    float h = 0.0f;
    for (int k = tid; k < K_EMB; k += 256) {
        float p = counts[k] / nv;
        h += p * __logf(p + 1e-10f);
    }
    red[tid] = h;
    __syncthreads();
    for (int s = 128; s > 0; s >>= 1) {
        if (tid < s) red[tid] += red[tid + s];
        __syncthreads();
    }
    if (tid == 0) {
        float perp    = __expf(-red[0]);
        float loss_vq = scalars[1] / (nv * (float)DIM);
        out[TL_OFF]   = loss_vq + 0.01f * (-__logf(perp + 1e-10f));
        out[PERP_OFF] = perp;
    }
}

// ---------------------------------------------------------------------------
extern "C" void kernel_launch(void* const* d_in, const int* in_sizes, int n_in,
                              void* d_out, int out_size, void* d_ws, size_t ws_size,
                              hipStream_t stream) {
    const float* inp = (const float*)d_in[0];   // [8,2048,64]
    const float* cb  = (const float*)d_in[1];   // [8192,64]
    const float* gum = (const float*)d_in[2];   // [16384,10]
    float* out = (float*)d_out;
    char*  ws  = (char*)d_ws;

    // workspace layout (needs ~6.2 MB)
    bf16*  cbHi    = (bf16*)(ws);                               // 1 MB
    bf16*  cbLo    = (bf16*)(ws + (1u << 20));                  // 1 MB
    bf16*  inHi    = (bf16*)(ws + (2u << 20));                  // 2 MB
    bf16*  inLo    = (bf16*)(ws + (4u << 20));                  // 2 MB
    float* xx      = (float*)(ws + (6u << 20));                 // 64 KB
    float* ee      = (float*)(ws + (6u << 20) + (64u << 10));   // 32 KB
    float* counts  = (float*)(ws + (6u << 20) + (96u << 10));   // 32 KB
    float* scalars = (float*)(ws + (6u << 20) + (128u << 10));  // 2 floats

    vq_init<<<(K_EMB + 255) / 256, 256, 0, stream>>>(counts, scalars);
    vq_stats<<<(N_ROWS + K_EMB + 255) / 256, 256, 0, stream>>>(inp, cb, xx, ee);
    vq_pack<<<(CODE_TILES * 64 + 255) / 256, 256, 0, stream>>>(cb, cbHi, cbLo, CODE_TILES, 1);
    vq_pack<<<(ROW_TILES * 64 + 255) / 256, 256, 0, stream>>>(inp, inHi, inLo, ROW_TILES, 0);
    vq_main<<<ROW_TILES / 8, 256, 0, stream>>>(inp, cb, gum, inHi, inLo, cbHi, cbLo,
                                               xx, ee, counts, scalars, out);
    vq_finalize<<<1, 256, 0, stream>>>(counts, scalars, out);
}
